// MixtureOfIntegrators_81990925681036
// MI455X (gfx1250) — compile-verified, matrix operand order checked by reading, not run on hardware
//
#include <hip/hip_runtime.h>
#include <hip/hip_bf16.h>
#include <math.h>

typedef __bf16 bf16;
typedef __attribute__((ext_vector_type(16))) __bf16 v16bf;
typedef __attribute__((ext_vector_type(8)))  __bf16 v8bf;
typedef __attribute__((ext_vector_type(8)))  float  v8f;

constexpr int B_   = 8192;
constexpr int HIDD = 2048;
constexpr int OUTD = 1024;
constexpr int NE   = 8;
constexpr int CTX  = 4096;       // HIDD + 2*OUTD
constexpr int CH   = 64;         // controller hidden
constexpr float DTC = 0.1f;
constexpr int TILE_M = 128;
constexpr int MAX_TILES = 2 * B_ / TILE_M + NE;   // 136 worst case
constexpr int KT1 = CTX / 32;                     // 128 k-tiles in GEMM1

// ---- output layout (floats) ----
constexpr size_t XO  = 0;
constexpr size_t VO  = (size_t)B_ * OUTD;
constexpr size_t RPO = 2 * (size_t)B_ * OUTD;
constexpr size_t IXO = RPO + (size_t)B_ * NE;
constexpr size_t TPO = IXO + (size_t)B_ * 2;

// ---- workspace layout (bytes) ----
constexpr size_t WS_TOPIDX = 128;                          // int[2B]
constexpr size_t WS_TOPW   = WS_TOPIDX + 8 * B_;           // float[2B]
constexpr size_t WS_ATOK   = WS_TOPW   + 8 * B_;           // int[2B]
constexpr size_t WS_AW     = WS_ATOK   + 8 * B_;           // float[2B]
constexpr size_t WS_W1S    = WS_AW     + 8 * B_;           // bf16 swizzled W1
constexpr size_t N_W1S     = (size_t)NE * KT1 * 4 * 512;   // 2,097,152 elems
constexpr size_t WS_W2S    = WS_W1S + 2 * N_W1S;
constexpr size_t N_W2S     = (size_t)NE * 64 * 3 * 2 * 512; // 1,572,864 elems
constexpr size_t WS_HID    = WS_W2S + 2 * N_W2S;           // bf16[2B*64]

__device__ __forceinline__ bf16 f2bf(float f) { return (bf16)f; }
__device__ __forceinline__ float sigm(float x) { return 1.f / (1.f + expf(-x)); }
__device__ __forceinline__ float sftp(float x) { return x > 20.f ? x : log1pf(expf(x)); }

#define WMMA_BF16(Afrag, Bfrag, Cfrag) \
  __builtin_amdgcn_wmma_f32_16x16x32_bf16(false, (Afrag), false, (Bfrag), (short)0, (Cfrag), false, false)

// A fragment from 32 contiguous f32 (two 32B chunks: k=[0,8)+[16,24) for half=0 layout)
__device__ __forceinline__ v16bf loadA_f32(const float* p) {
  const float4* p0 = (const float4*)p;
  const float4* p1 = (const float4*)(p + 16);
  float4 f0 = p0[0], f1 = p0[1];
  float4 f2 = p1[0], f3 = p1[1];
  v16bf a;
  a[0]=f2bf(f0.x);  a[1]=f2bf(f0.y);  a[2]=f2bf(f0.z);  a[3]=f2bf(f0.w);
  a[4]=f2bf(f1.x);  a[5]=f2bf(f1.y);  a[6]=f2bf(f1.z);  a[7]=f2bf(f1.w);
  a[8]=f2bf(f2.x);  a[9]=f2bf(f2.y);  a[10]=f2bf(f2.z); a[11]=f2bf(f2.w);
  a[12]=f2bf(f3.x); a[13]=f2bf(f3.y); a[14]=f2bf(f3.z); a[15]=f2bf(f3.w);
  return a;
}

// ---------------- kernel 0: zero outputs + counts ----------------
__global__ void k_zero(float* out, int* meta) {
  size_t n = 2 * (size_t)B_ * OUTD;
  size_t stride = (size_t)gridDim.x * blockDim.x;
  for (size_t i = (size_t)blockIdx.x * blockDim.x + threadIdx.x; i < n; i += stride)
    out[i] = 0.f;
  if (blockIdx.x == 0 && threadIdx.x < 8) meta[threadIdx.x] = 0;
}

// ---------------- kernel 1: router (1 wave per token) ----------------
__global__ __launch_bounds__(256) void k_router(const float* __restrict__ h,
                                                const float* __restrict__ rw,
                                                const float* __restrict__ rb,
                                                float* __restrict__ out,
                                                int* __restrict__ meta,
                                                int* __restrict__ top_idx,
                                                float* __restrict__ top_w) {
  int wave = threadIdx.x >> 5, lane = threadIdx.x & 31;
  int t = blockIdx.x * 8 + wave;
  if (t >= B_) return;
  float acc[NE];
#pragma unroll
  for (int e = 0; e < NE; ++e) acc[e] = 0.f;
  const float* hr = h + (size_t)t * HIDD;
  for (int k = lane; k < HIDD; k += 32) {
    float hv = hr[k];
    const float4* rp = (const float4*)(rw + (size_t)k * NE);
    float4 w0 = rp[0], w1 = rp[1];
    acc[0] += hv * w0.x; acc[1] += hv * w0.y; acc[2] += hv * w0.z; acc[3] += hv * w0.w;
    acc[4] += hv * w1.x; acc[5] += hv * w1.y; acc[6] += hv * w1.z; acc[7] += hv * w1.w;
  }
#pragma unroll
  for (int e = 0; e < NE; ++e)
    for (int s = 16; s > 0; s >>= 1) acc[e] += __shfl_down(acc[e], s);
  if (lane == 0) {
    float lg[NE], pr[NE];
    float m = -1e30f;
#pragma unroll
    for (int e = 0; e < NE; ++e) { lg[e] = acc[e] + rb[e]; m = fmaxf(m, lg[e]); }
    float s = 0.f;
#pragma unroll
    for (int e = 0; e < NE; ++e) { pr[e] = expf(lg[e] - m); s += pr[e]; }
    float inv = 1.f / s;
#pragma unroll
    for (int e = 0; e < NE; ++e) { pr[e] *= inv; out[RPO + (size_t)t * NE + e] = pr[e]; }
    int i1 = 0;
#pragma unroll
    for (int e = 1; e < NE; ++e) if (pr[e] > pr[i1]) i1 = e;
    int i2 = (i1 == 0) ? 1 : 0;
#pragma unroll
    for (int e = 0; e < NE; ++e) if (e != i1 && pr[e] > pr[i2]) i2 = e;
    float p1 = pr[i1], p2 = pr[i2];
    float rs = 1.f / (p1 + p2);
    out[IXO + 2 * t + 0] = (float)i1;
    out[IXO + 2 * t + 1] = (float)i2;
    out[TPO + 2 * t + 0] = p1 * rs;
    out[TPO + 2 * t + 1] = p2 * rs;
    top_idx[2 * t + 0] = i1; top_idx[2 * t + 1] = i2;
    top_w[2 * t + 0] = p1 * rs; top_w[2 * t + 1] = p2 * rs;
    atomicAdd(&meta[i1], 1);
    atomicAdd(&meta[i2], 1);
  }
}

// ---------------- kernel 2: prefix scan (tiny) ----------------
__global__ void k_scan(int* meta) {
  if (threadIdx.x == 0 && blockIdx.x == 0) {
    int* counts = meta; int* offs = meta + 8; int* cursor = meta + 17;
    int acc = 0;
    for (int i = 0; i < NE; ++i) { offs[i] = acc; cursor[i] = acc; acc += counts[i]; }
    offs[NE] = acc;
  }
}

// ---------------- kernel 3: scatter assignments ----------------
__global__ void k_scatter(const int* __restrict__ top_idx, const float* __restrict__ top_w,
                          int* __restrict__ meta, int* __restrict__ atok, float* __restrict__ aw) {
  int t = blockIdx.x * blockDim.x + threadIdx.x;
  if (t >= B_) return;
  int* cursor = meta + 17;
#pragma unroll
  for (int k = 0; k < 2; ++k) {
    int e = top_idx[2 * t + k];
    int slot = atomicAdd(&cursor[e], 1);
    atok[slot] = t;
    aw[slot] = top_w[2 * t + k];
  }
}

// ---------------- kernel 4: swizzle weights to fragment-ready bf16 ----------------
// W1s: [((e*128 + kt)*4 + nt)*512 + lane*16 + m], frag elem m -> k_local=(m>>3)*16+half*8+(m&7), n=nt*16+lane%16
// W2s: [(((e*64 + dc)*3 + g)*2 + ks)*512 + lane*16 + m], k=ks*32+k_local, col=g*1024+dc*16+lane%16
__global__ void k_swizzle(const float* __restrict__ W1, const float* __restrict__ W2,
                          bf16* __restrict__ W1s, bf16* __restrict__ W2s) {
  size_t stride = (size_t)gridDim.x * blockDim.x;
  for (size_t i = (size_t)blockIdx.x * blockDim.x + threadIdx.x; i < N_W1S + N_W2S; i += stride) {
    if (i < N_W1S) {
      int m = (int)(i & 15), lane = (int)((i >> 4) & 31);
      size_t rest = i >> 9;
      int nt = (int)(rest & 3);
      int kt = (int)((rest >> 2) & 127);
      int e  = (int)(rest >> 9);
      int half = lane >> 4;
      int n  = nt * 16 + (lane & 15);
      int kl = (m >> 3) * 16 + half * 8 + (m & 7);
      W1s[i] = f2bf(W1[((size_t)e * CTX + kt * 32 + kl) * CH + n]);
    } else {
      size_t j = i - N_W1S;
      int m = (int)(j & 15), lane = (int)((j >> 4) & 31);
      size_t rest = j >> 9;
      int ks = (int)(rest & 1);
      size_t r3 = rest >> 1;
      int g = (int)(r3 % 3);
      size_t r4 = r3 / 3;
      int dc = (int)(r4 & 63);
      int e  = (int)(r4 >> 6);
      int half = lane >> 4;
      int kl = (m >> 3) * 16 + half * 8 + (m & 7);
      int k  = ks * 32 + kl;
      int col = g * OUTD + dc * 16 + (lane & 15);
      W2s[j] = f2bf(W2[((size_t)e * CH + k) * (3 * OUTD) + col]);
    }
  }
}

// ---------------- block -> (expert, tile) mapping ----------------
__device__ __forceinline__ bool map_block(const int* offs, int b,
                                          int& e, int& row0, int& cnt, int& segbase) {
  int acc = 0;
  for (int i = 0; i < NE; ++i) {
    int c = offs[i + 1] - offs[i];
    int nt = (c + TILE_M - 1) / TILE_M;
    if (b < acc + nt) { e = i; row0 = (b - acc) * TILE_M; cnt = c; segbase = offs[i]; return true; }
    acc += nt;
  }
  return false;
}

// ---------------- kernel 5: GEMM1  hid = relu(ctx @ W1[e] + b1[e]) ----------------
// 4 waves/block, 32 rows per wave (two A frags) -> each B frag reused twice.
__global__ __launch_bounds__(128) void k_gemm1(const float* __restrict__ h,
                                               const float* __restrict__ x,
                                               const float* __restrict__ v,
                                               const bf16* __restrict__ W1s,
                                               const float* __restrict__ b1,
                                               const int* __restrict__ meta,
                                               const int* __restrict__ atok,
                                               bf16* __restrict__ hid) {
  const int* offs = meta + 8;
  int e, row0, cnt, segbase;
  if (!map_block(offs, blockIdx.x, e, row0, cnt, segbase)) return;
  int tid = threadIdx.x, lane = tid & 31, wave = tid >> 5;   // wave 0..3
  int half = lane >> 4, mloc = lane & 15;
  int rbase = row0 + wave * 32;

  int rowA0 = rbase + mloc;
  int rowA1 = rbase + 16 + mloc;
  int tok0 = atok[segbase + (rowA0 < cnt ? rowA0 : cnt - 1)];
  int tok1 = atok[segbase + (rowA1 < cnt ? rowA1 : cnt - 1)];

  v8f z = {0.f,0.f,0.f,0.f,0.f,0.f,0.f,0.f};
  v8f acc0[4] = {z, z, z, z};
  v8f acc1[4] = {z, z, z, z};

  const bf16* wb = W1s + (size_t)e * (KT1 * 4 * 512) + lane * 16;

  const float* bp0[3];
  const float* bp1[3];
  bp0[0] = h + (size_t)tok0 * HIDD + half * 8;
  bp0[1] = x + (size_t)tok0 * OUTD + half * 8;
  bp0[2] = v + (size_t)tok0 * OUTD + half * 8;
  bp1[0] = h + (size_t)tok1 * HIDD + half * 8;
  bp1[1] = x + (size_t)tok1 * OUTD + half * 8;
  bp1[2] = v + (size_t)tok1 * OUTD + half * 8;
  const int nks[3] = {HIDD / 32, OUTD / 32, OUTD / 32};

  int kt = 0;
#pragma unroll
  for (int ph = 0; ph < 3; ++ph) {
    const float* p0 = bp0[ph];
    const float* p1 = bp1[ph];
    int nk = nks[ph];
#pragma unroll 1
    for (int ki = 0; ki < nk; ++ki, ++kt) {
      v16bf a0 = loadA_f32(p0 + ki * 32);
      v16bf a1 = loadA_f32(p1 + ki * 32);
      const bf16* wkt = wb + (size_t)kt * (4 * 512);
#pragma unroll
      for (int nt = 0; nt < 4; ++nt) {
        v16bf bb = *(const v16bf*)(wkt + nt * 512);
        acc0[nt] = WMMA_BF16(a0, bb, acc0[nt]);
        acc1[nt] = WMMA_BF16(a1, bb, acc1[nt]);
      }
    }
  }

  // epilogue: bias + relu -> hid (bf16). D layout: row = r + 8*half, col = lane%16.
  int mhi = half * 8;
#pragma unroll
  for (int nt = 0; nt < 4; ++nt) {
    int col = nt * 16 + mloc;
    float bias = b1[e * CH + col];
#pragma unroll
    for (int mt = 0; mt < 2; ++mt) {
#pragma unroll
      for (int r = 0; r < 8; ++r) {
        int grow = rbase + mt * 16 + mhi + r;
        if (grow < cnt) {
          float val = (mt ? acc1[nt][r] : acc0[nt][r]) + bias;
          hid[(size_t)(segbase + grow) * CH + col] = f2bf(fmaxf(val, 0.f));
        }
      }
    }
  }
}

// ---------------- kernel 6: GEMM2 + activations + integrator + combine ----------------
__global__ __launch_bounds__(128) void k_gemm2(const float* __restrict__ x,
                                               const float* __restrict__ v,
                                               const float* __restrict__ mu,
                                               const bf16* __restrict__ W2s,
                                               const float* __restrict__ b2,
                                               const int* __restrict__ meta,
                                               const int* __restrict__ atok,
                                               const float* __restrict__ aw,
                                               const bf16* __restrict__ hid,
                                               float* __restrict__ x_next,
                                               float* __restrict__ v_next) {
  const int* offs = meta + 8;
  int e, row0, cnt, segbase;
  if (!map_block(offs, blockIdx.x, e, row0, cnt, segbase)) return;
  int tid = threadIdx.x, lane = tid & 31, wave = tid >> 5;   // wave 0..3
  int half = lane >> 4, mloc = lane & 15;
  int mhi = half * 8;
  int rbase = row0 + wave * 32;

  // A fragments for two 16-row tiles (K=64 -> two frags each), built once from hid rows
  v16bf a00, a01, a10, a11;
  {
    int rowA0 = rbase + mloc;
    int rowA1 = rbase + 16 + mloc;
    const bf16* hr0 = hid + (size_t)(segbase + (rowA0 < cnt ? rowA0 : cnt - 1)) * CH;
    const bf16* hr1 = hid + (size_t)(segbase + (rowA1 < cnt ? rowA1 : cnt - 1)) * CH;
    v8bf c0 = *(const v8bf*)(hr0 + half * 8);
    v8bf c1 = *(const v8bf*)(hr0 + 16 + half * 8);
    v8bf c2 = *(const v8bf*)(hr0 + 32 + half * 8);
    v8bf c3 = *(const v8bf*)(hr0 + 48 + half * 8);
    v8bf d0 = *(const v8bf*)(hr1 + half * 8);
    v8bf d1 = *(const v8bf*)(hr1 + 16 + half * 8);
    v8bf d2 = *(const v8bf*)(hr1 + 32 + half * 8);
    v8bf d3 = *(const v8bf*)(hr1 + 48 + half * 8);
#pragma unroll
    for (int m = 0; m < 8; ++m) {
      a00[m] = c0[m]; a00[8 + m] = c1[m];
      a01[m] = c2[m]; a01[8 + m] = c3[m];
      a10[m] = d0[m]; a10[8 + m] = d1[m];
      a11[m] = d2[m]; a11[8 + m] = d3[m];
    }
  }
  // per-D-row token/weight for both m tiles
  int tokD[2][8]; float wD[2][8];
#pragma unroll
  for (int mt = 0; mt < 2; ++mt) {
#pragma unroll
    for (int r = 0; r < 8; ++r) {
      int grow = rbase + mt * 16 + mhi + r;
      int rr = grow < cnt ? grow : cnt - 1;
      tokD[mt][r] = atok[segbase + rr];
      wD[mt][r] = aw[segbase + rr];
    }
  }

  v8f z = {0.f,0.f,0.f,0.f,0.f,0.f,0.f,0.f};
  const bf16* w2base = W2s + (size_t)e * (64 * 3 * 2 * 512) + lane * 16;
  const float* b2e = b2 + (size_t)e * 3 * OUTD;

#pragma unroll 2
  for (int dc = 0; dc < OUTD / 16; ++dc) {
    const bf16* wd = w2base + (size_t)dc * (3 * 2 * 512);
    v16bf b00 = *(const v16bf*)(wd + 0 * 512);
    v16bf b01 = *(const v16bf*)(wd + 1 * 512);
    v16bf b10 = *(const v16bf*)(wd + 2 * 512);
    v16bf b11 = *(const v16bf*)(wd + 3 * 512);
    v16bf b20 = *(const v16bf*)(wd + 4 * 512);
    v16bf b21 = *(const v16bf*)(wd + 5 * 512);
    v8f ca0 = z, cb0 = z, cg0 = z;
    v8f ca1 = z, cb1 = z, cg1 = z;
    ca0 = WMMA_BF16(a00, b00, ca0); ca0 = WMMA_BF16(a01, b01, ca0);
    ca1 = WMMA_BF16(a10, b00, ca1); ca1 = WMMA_BF16(a11, b01, ca1);
    cb0 = WMMA_BF16(a00, b10, cb0); cb0 = WMMA_BF16(a01, b11, cb0);
    cb1 = WMMA_BF16(a10, b10, cb1); cb1 = WMMA_BF16(a11, b11, cb1);
    cg0 = WMMA_BF16(a00, b20, cg0); cg0 = WMMA_BF16(a01, b21, cg0);
    cg1 = WMMA_BF16(a10, b20, cg1); cg1 = WMMA_BF16(a11, b21, cg1);

    int d = dc * 16 + mloc;
    float biasA = b2e[d];
    float biasB = b2e[OUTD + d];
    float biasG = b2e[2 * OUTD + d];
    float mud = mu[d];
#pragma unroll
    for (int mt = 0; mt < 2; ++mt) {
#pragma unroll
      for (int r = 0; r < 8; ++r) {
        int grow = rbase + mt * 16 + mhi + r;
        if (grow < cnt) {
          int tok = tokD[mt][r];
          float w = wD[mt][r];
          float va = (mt ? ca1[r] : ca0[r]) + biasA;
          float vb = (mt ? cb1[r] : cb0[r]) + biasB;
          float vg = (mt ? cg1[r] : cg0[r]) + biasG;
          float alpha = sigm(va);
          float beta  = sftp(vb);
          float gate  = sigm(vg);
          float xv = x[(size_t)tok * OUTD + d];
          float vv = v[(size_t)tok * OUTD + d];
          float err = xv - mud;
          float vn = alpha * vv - beta * err;
          float xn = xv + DTC * gate * vn;
          atomicAdd(&x_next[(size_t)tok * OUTD + d], w * xn);
          atomicAdd(&v_next[(size_t)tok * OUTD + d], w * vn);
        }
      }
    }
  }
}

// ---------------- launcher ----------------
extern "C" void kernel_launch(void* const* d_in, const int* in_sizes, int n_in,
                              void* d_out, int out_size, void* d_ws, size_t ws_size,
                              hipStream_t stream) {
  const float* h  = (const float*)d_in[0];
  const float* x  = (const float*)d_in[1];
  const float* v  = (const float*)d_in[2];
  const float* rw = (const float*)d_in[3];
  const float* rb = (const float*)d_in[4];
  const float* mu = (const float*)d_in[5];
  const float* W1 = (const float*)d_in[6];
  const float* b1 = (const float*)d_in[7];
  const float* W2 = (const float*)d_in[8];
  const float* b2 = (const float*)d_in[9];
  float* out = (float*)d_out;

  char* ws = (char*)d_ws;
  int*   meta    = (int*)ws;
  int*   top_idx = (int*)(ws + WS_TOPIDX);
  float* top_w   = (float*)(ws + WS_TOPW);
  int*   atok    = (int*)(ws + WS_ATOK);
  float* aw      = (float*)(ws + WS_AW);
  bf16*  W1s     = (bf16*)(ws + WS_W1S);
  bf16*  W2s     = (bf16*)(ws + WS_W2S);
  bf16*  hid     = (bf16*)(ws + WS_HID);

  k_zero<<<4096, 256, 0, stream>>>(out, meta);
  k_router<<<B_ / 8, 256, 0, stream>>>(h, rw, rb, out, meta, top_idx, top_w);
  k_scan<<<1, 32, 0, stream>>>(meta);
  k_scatter<<<B_ / 256, 256, 0, stream>>>(top_idx, top_w, meta, atok, aw);
  k_swizzle<<<14336, 256, 0, stream>>>(W1, W2, W1s, W2s);
  k_gemm1<<<MAX_TILES, 128, 0, stream>>>(h, x, v, W1s, b1, meta, atok, hid);
  k_gemm2<<<MAX_TILES, 128, 0, stream>>>(x, v, mu, W2s, b2, meta, atok, aw, hid,
                                         out + XO, out + VO);
}
